// MLA_19155554140881
// MI455X (gfx1250) — compile-verified, matrix operand order checked by reading, not run on hardware
//
#include <hip/hip_runtime.h>
#include <hip/hip_bf16.h>
#include <stdint.h>

typedef __bf16 bf16;
typedef bf16 bf16x8  __attribute__((ext_vector_type(8)));
typedef bf16 bf16x16 __attribute__((ext_vector_type(16)));
typedef float v8f    __attribute__((ext_vector_type(8)));

#define B_    4
#define T_    2048
#define DIM_  1024
#define H_    16
#define NOPE_ 128
#define ROPE_ 64
#define QKD_  192      // NOPE + ROPE
#define VDIM_ 128
#define KVR_  512
#define ROWS_ (B_*T_)  // 8192

union Frag { bf16x16 v; bf16x8 h[2]; };

// Issue one 16-byte async memory->LDS copy (gfx1250 GLOBAL_LOAD_ASYNC_TO_LDS_B128).
__device__ __forceinline__ void async_copy_b128(const void* gptr, void* lptr) {
  uint32_t lds = (uint32_t)(uintptr_t)lptr;     // low 32 bits of flat addr == LDS offset
  uint64_t ga  = (uint64_t)(uintptr_t)gptr;
  asm volatile("global_load_async_to_lds_b128 %0, %1, off"
               :: "v"(lds), "v"(ga) : "memory");
}
__device__ __forceinline__ void async_wait0() {
  asm volatile("s_wait_asynccnt 0x0" ::: "memory");
}

// ---------------------------------------------------------------- convert
__global__ void cvt_f32_bf16(const float* __restrict__ in, bf16* __restrict__ out, int n) {
  int i = blockIdx.x * blockDim.x + threadIdx.x;
  if (i < n) out[i] = (bf16)in[i];
}

// ---------------------------------------------------------------- scatter epilogue helper
__device__ __forceinline__ void store_elem(int mode, void* C0, void* C1,
                                           int row, int col, int N, float v) {
  if (mode == 0) {
    ((bf16*)C0)[(size_t)row*N + col] = (bf16)v;
  } else if (mode == 1) {
    ((float*)C0)[(size_t)row*N + col] = v;
  } else { // MLA kv scatter: head h = col>>8, c = col&255
    int h = col >> 8, cc = col & 255;
    if (cc < NOPE_)
      ((bf16*)C0)[(size_t)row*(H_*QKD_)  + h*QKD_  + cc] = (bf16)v;
    else
      ((bf16*)C1)[(size_t)row*(H_*VDIM_) + h*VDIM_ + (cc-NOPE_)] = (bf16)v;
  }
}

// ---------------------------------------------------------------- GEMM 128x128 tile
// C[M,N] = A[M,K](bf16,row) @ B[N,K](bf16,row)^T + bias.  M%128==0, N%128==0, K%32==0.
// Double-buffered LDS fed by async global->LDS copies; 8 waves in 4x2 grid,
// each wave computes 32x64 via 8 WMMA accumulators.
__global__ __launch_bounds__(256)
void gemm_wmma128(const bf16* __restrict__ A, const bf16* __restrict__ Bw,
                  const float* __restrict__ bias, void* __restrict__ C0,
                  void* __restrict__ C1, int M, int N, int K, int mode)
{
  __shared__ __align__(16) bf16 As[2][128][40];
  __shared__ __align__(16) bf16 Bs[2][128][40];
  const int t = threadIdx.x;
  const int lane = t & 31, w = t >> 5;
  const int wm = w >> 1, wn = w & 1;          // 4x2 wave grid
  const int r = lane & 15, hf = lane >> 4;
  const int m0 = blockIdx.y * 128, n0 = blockIdx.x * 128;

  auto issue = [&](int k0, int buf) {
    #pragma unroll
    for (int i = 0; i < 2; ++i) {
      int c = t + i * 256;                    // 512 chunks of 8 bf16 per matrix
      int row = c >> 2, ch = c & 3;
      async_copy_b128(A  + (size_t)(m0+row)*K + k0 + ch*8, &As[buf][row][ch*8]);
      async_copy_b128(Bw + (size_t)(n0+row)*K + k0 + ch*8, &Bs[buf][row][ch*8]);
    }
  };

  v8f acc[2][4];
  #pragma unroll
  for (int mi = 0; mi < 2; ++mi)
    #pragma unroll
    for (int ni = 0; ni < 4; ++ni)
      #pragma unroll
      for (int g = 0; g < 8; ++g) acc[mi][ni][g] = 0.f;

  const int ntiles = K >> 5;
  issue(0, 0);
  async_wait0();
  __syncthreads();

  for (int i = 0; i < ntiles; ++i) {
    const int buf = i & 1;
    if (i + 1 < ntiles) issue((i + 1) << 5, buf ^ 1);
    if (i + 2 < ntiles)  // warm L2 for the tile after next
      __builtin_prefetch(A + (size_t)(m0 + (t >> 1)) * K + ((i + 2) << 5), 0, 1);

    Frag a[2], b[4];
    #pragma unroll
    for (int mi = 0; mi < 2; ++mi) {
      a[mi].h[0] = *(const bf16x8*)&As[buf][wm*32 + mi*16 + r][hf*8];
      a[mi].h[1] = *(const bf16x8*)&As[buf][wm*32 + mi*16 + r][16 + hf*8];
    }
    #pragma unroll
    for (int ni = 0; ni < 4; ++ni) {
      b[ni].h[0] = *(const bf16x8*)&Bs[buf][wn*64 + ni*16 + r][hf*8];
      b[ni].h[1] = *(const bf16x8*)&Bs[buf][wn*64 + ni*16 + r][16 + hf*8];
    }
    #pragma unroll
    for (int mi = 0; mi < 2; ++mi)
      #pragma unroll
      for (int ni = 0; ni < 4; ++ni)
        acc[mi][ni] = __builtin_amdgcn_wmma_f32_16x16x32_bf16(
            false, a[mi].v, false, b[ni].v, (short)0, acc[mi][ni], false, false);

    async_wait0();      // own async loads for buf^1 landed
    __syncthreads();    // everyone done reading buf / loading buf^1
  }

  #pragma unroll
  for (int mi = 0; mi < 2; ++mi)
    #pragma unroll
    for (int ni = 0; ni < 4; ++ni)
      #pragma unroll
      for (int g = 0; g < 8; ++g) {
        const int row = m0 + wm*32 + mi*16 + hf*8 + g;
        const int col = n0 + wn*64 + ni*16 + r;
        store_elem(mode, C0, C1, row, col, N, acc[mi][ni][g] + bias[col]);
      }
}

// ---------------------------------------------------------------- GEMM 64x64 tile (for N%128!=0)
__global__ __launch_bounds__(256)
void gemm_wmma(const bf16* __restrict__ A, const bf16* __restrict__ Bw,
               const float* __restrict__ bias, void* __restrict__ C0,
               void* __restrict__ C1, int M, int N, int K, int mode)
{
  __shared__ __align__(16) bf16 As[64][40];
  __shared__ __align__(16) bf16 Bs[64][40];
  const int t = threadIdx.x;
  const int lane = t & 31, w = t >> 5;
  const int wm = w >> 1, wn = w & 1;
  const int r = lane & 15, hf = lane >> 4;
  const int m0 = blockIdx.y * 64, n0 = blockIdx.x * 64;
  const int arow = t >> 2, ach = t & 3;

  v8f acc0 = {0.f,0.f,0.f,0.f,0.f,0.f,0.f,0.f};
  v8f acc1 = {0.f,0.f,0.f,0.f,0.f,0.f,0.f,0.f};

  for (int k0 = 0; k0 < K; k0 += 32) {
    __syncthreads();
    *(uint4*)&As[arow][ach*8] = *(const uint4*)(A  + (size_t)(m0+arow)*K + k0 + ach*8);
    *(uint4*)&Bs[arow][ach*8] = *(const uint4*)(Bw + (size_t)(n0+arow)*K + k0 + ach*8);
    __syncthreads();
    Frag a, b0, b1;
    a.h[0]  = *(const bf16x8*)&As[wm*16 + r][hf*8];
    a.h[1]  = *(const bf16x8*)&As[wm*16 + r][16 + hf*8];
    b0.h[0] = *(const bf16x8*)&Bs[wn*32 + r][hf*8];
    b0.h[1] = *(const bf16x8*)&Bs[wn*32 + r][16 + hf*8];
    b1.h[0] = *(const bf16x8*)&Bs[wn*32 + 16 + r][hf*8];
    b1.h[1] = *(const bf16x8*)&Bs[wn*32 + 16 + r][16 + hf*8];
    acc0 = __builtin_amdgcn_wmma_f32_16x16x32_bf16(false, a.v, false, b0.v, (short)0, acc0, false, false);
    acc1 = __builtin_amdgcn_wmma_f32_16x16x32_bf16(false, a.v, false, b1.v, (short)0, acc1, false, false);
  }

  #pragma unroll
  for (int g = 0; g < 8; ++g) {
    const int row = m0 + wm*16 + hf*8 + g;
    const int c0  = n0 + wn*32 + r;
    store_elem(mode, C0, C1, row, c0,      N, acc0[g] + bias[c0]);
    store_elem(mode, C0, C1, row, c0 + 16, N, acc1[g] + bias[c0 + 16]);
  }
}

// ---------------------------------------------------------------- RoPE on q (in place, bf16)
__global__ void rope_q(bf16* __restrict__ q, const float* __restrict__ fc,
                       const float* __restrict__ fs) {
  int idx = blockIdx.x * blockDim.x + threadIdx.x;
  if (idx >= ROWS_ * H_ * (ROPE_/2)) return;
  int i   = idx & 31;
  int h   = (idx >> 5) & 15;
  int row = idx >> 9;
  int tp  = row & (T_ - 1);
  size_t base = (size_t)row*(H_*QKD_) + h*QKD_ + NOPE_ + 2*i;
  float xr = (float)q[base], xi = (float)q[base+1];
  float c = fc[tp*32 + i], s = fs[tp*32 + i];
  q[base]   = (bf16)(xr*c - xi*s);
  q[base+1] = (bf16)(xr*s + xi*c);
}

// ---------------------------------------------------------------- rms_norm(kv_lat) + k_rope broadcast
__global__ __launch_bounds__(256)
void kv_post(const bf16* __restrict__ kva, bf16* __restrict__ kvn,
             bf16* __restrict__ kb, const float* __restrict__ fc,
             const float* __restrict__ fs)
{
  __shared__ float red[8];
  const int row = blockIdx.x, t = threadIdx.x;
  const int tp = row & (T_ - 1);
  const bf16* src = kva + (size_t)row * (KVR_ + ROPE_);
  float v0 = (float)src[t], v1 = (float)src[t + 256];
  float ss = v0*v0 + v1*v1;
  #pragma unroll
  for (int m = 1; m < 32; m <<= 1) ss += __shfl_xor(ss, m, 32);
  if ((t & 31) == 0) red[t >> 5] = ss;
  __syncthreads();
  float tot = 0.f;
  #pragma unroll
  for (int i = 0; i < 8; ++i) tot += red[i];
  float rn = rsqrtf(tot * (1.0f/512.0f) + 1e-6f);
  bf16* dn = kvn + (size_t)row * KVR_;
  dn[t]       = (bf16)(v0 * rn);
  dn[t + 256] = (bf16)(v1 * rn);
  if (t < 32) {
    float xr = (float)src[KVR_ + 2*t], xi = (float)src[KVR_ + 2*t + 1];
    float c = fc[tp*32 + t], s = fs[tp*32 + t];
    float yr = xr*c - xi*s, yi = xr*s + xi*c;
    bf16* kr = kb + (size_t)row * (H_*QKD_);
    #pragma unroll
    for (int h = 0; h < H_; ++h) {
      kr[h*QKD_ + NOPE_ + 2*t]     = (bf16)yr;
      kr[h*QKD_ + NOPE_ + 2*t + 1] = (bf16)yi;
    }
  }
}

// ---------------------------------------------------------------- flash attention
// 128 q-rows/block, 8 waves x 16 rows; K tile staged via async global->LDS.
__global__ __launch_bounds__(256)
void attn_wmma(const bf16* __restrict__ Q, const bf16* __restrict__ Kk,
               const bf16* __restrict__ V, bf16* __restrict__ Oo)
{
  __shared__ __align__(16) bf16 Ks[64][200];     // 64 keys x 192 (+pad)
  __shared__ __align__(16) bf16 Vt[128][72];     // vdim x keys (+pad)
  __shared__ __align__(16) bf16 Ps[8][16][72];   // per-wave P staging
  const int t = threadIdx.x, lane = t & 31, w = t >> 5;
  const int r = lane & 15, hf = lane >> 4;
  const int b = blockIdx.z, h = blockIdx.y, qt = blockIdx.x;
  const size_t rowQ0 = (size_t)b * T_ + (size_t)qt * 128;
  const float scale = 0.07216878364870322f;      // 1/sqrt(192)

  Frag aq[6];
  {
    const size_t qbase = (rowQ0 + w*16 + r) * (size_t)(H_*QKD_) + h*QKD_;
    #pragma unroll
    for (int ks = 0; ks < 6; ++ks) {
      aq[ks].h[0] = *(const bf16x8*)(Q + qbase + ks*32 + hf*8);
      aq[ks].h[1] = *(const bf16x8*)(Q + qbase + ks*32 + 16 + hf*8);
    }
  }

  v8f o[8];
  #pragma unroll
  for (int n = 0; n < 8; ++n)
    #pragma unroll
    for (int g = 0; g < 8; ++g) o[n][g] = 0.f;
  float mrow[8], lrow[8];
  #pragma unroll
  for (int g = 0; g < 8; ++g) { mrow[g] = -3.0e38f; lrow[g] = 0.f; }

  const int nkt = qt*2 + 2;   // causal: key tiles 0 .. qt*2+1
  for (int kt = 0; kt < nkt; ++kt) {
    __syncthreads();
    {
      const size_t krow0 = (size_t)b * T_ + (size_t)kt * 64;
      for (int c = t; c < 1536; c += 256) {          // K tile 64x192 via async
        int rr = c / 24, ch = c % 24;
        async_copy_b128(Kk + (krow0+rr)*(size_t)(H_*QKD_) + h*QKD_ + ch*8,
                        &Ks[rr][ch*8]);
      }
      for (int c = t; c < 1024; c += 256) {          // V tile 64x128, transposed
        int ss = c >> 4, ch = c & 15;
        uint4 raw = *(const uint4*)(V + (krow0+ss)*(size_t)(H_*VDIM_) + h*VDIM_ + ch*8);
        const bf16* pv = (const bf16*)&raw;
        #pragma unroll
        for (int j = 0; j < 8; ++j) Vt[ch*8+j][ss] = pv[j];
      }
    }
    async_wait0();
    __syncthreads();

    v8f s[4];
    #pragma unroll
    for (int n = 0; n < 4; ++n)
      #pragma unroll
      for (int g = 0; g < 8; ++g) s[n][g] = 0.f;
    #pragma unroll
    for (int ks = 0; ks < 6; ++ks) {
      #pragma unroll
      for (int n = 0; n < 4; ++n) {
        Frag bk;
        bk.h[0] = *(const bf16x8*)&Ks[n*16 + r][ks*32 + hf*8];
        bk.h[1] = *(const bf16x8*)&Ks[n*16 + r][ks*32 + 16 + hf*8];
        s[n] = __builtin_amdgcn_wmma_f32_16x16x32_bf16(false, aq[ks].v, false, bk.v,
                                                       (short)0, s[n], false, false);
      }
    }

    const int qrow_base = qt*128 + w*16 + hf*8;
    #pragma unroll
    for (int g = 0; g < 8; ++g) {
      const int qrow = qrow_base + g;
      float mx = -3.0e38f;
      #pragma unroll
      for (int n = 0; n < 4; ++n) {
        int col = kt*64 + n*16 + r;
        float vs = s[n][g] * scale;
        if (col > qrow) vs = -3.0e38f;
        s[n][g] = vs;
        mx = fmaxf(mx, vs);
      }
      #pragma unroll
      for (int m = 1; m < 16; m <<= 1) mx = fmaxf(mx, __shfl_xor(mx, m, 16));
      float mnew  = fmaxf(mrow[g], mx);
      float fcorr = __expf(mrow[g] - mnew);
      float rs = 0.f;
      #pragma unroll
      for (int n = 0; n < 4; ++n) {
        float p = __expf(s[n][g] - mnew);
        Ps[w][hf*8 + g][n*16 + r] = (bf16)p;
        rs += p;
      }
      #pragma unroll
      for (int m = 1; m < 16; m <<= 1) rs += __shfl_xor(rs, m, 16);
      lrow[g] = lrow[g] * fcorr + rs;
      mrow[g] = mnew;
      #pragma unroll
      for (int n2 = 0; n2 < 8; ++n2) o[n2][g] *= fcorr;
    }

    #pragma unroll
    for (int kk = 0; kk < 2; ++kk) {
      Frag ap;
      ap.h[0] = *(const bf16x8*)&Ps[w][r][kk*32 + hf*8];
      ap.h[1] = *(const bf16x8*)&Ps[w][r][kk*32 + 16 + hf*8];
      #pragma unroll
      for (int n2 = 0; n2 < 8; ++n2) {
        Frag bv;
        bv.h[0] = *(const bf16x8*)&Vt[n2*16 + r][kk*32 + hf*8];
        bv.h[1] = *(const bf16x8*)&Vt[n2*16 + r][kk*32 + 16 + hf*8];
        o[n2] = __builtin_amdgcn_wmma_f32_16x16x32_bf16(false, ap.v, false, bv.v,
                                                        (short)0, o[n2], false, false);
      }
    }
  }

  #pragma unroll
  for (int g = 0; g < 8; ++g) {
    float inv = 1.0f / lrow[g];
    size_t row = rowQ0 + w*16 + hf*8 + g;
    #pragma unroll
    for (int n2 = 0; n2 < 8; ++n2)
      Oo[row*(size_t)(H_*VDIM_) + h*VDIM_ + n2*16 + r] = (bf16)(o[n2][g] * inv);
  }
}

// ---------------------------------------------------------------- host
extern "C" void kernel_launch(void* const* d_in, const int* in_sizes, int n_in,
                              void* d_out, int out_size, void* d_ws, size_t ws_size,
                              hipStream_t stream)
{
  (void)in_sizes; (void)n_in; (void)out_size; (void)ws_size;
  const float* x       = (const float*)d_in[0];
  const float* wq_w    = (const float*)d_in[1];
  const float* wq_b    = (const float*)d_in[2];
  const float* wkv_a_w = (const float*)d_in[3];
  const float* wkv_a_b = (const float*)d_in[4];
  const float* wkv_b_w = (const float*)d_in[5];
  const float* wkv_b_b = (const float*)d_in[6];
  const float* wo_w    = (const float*)d_in[7];
  const float* wo_b    = (const float*)d_in[8];
  const float* fcos    = (const float*)d_in[9];
  const float* fsin    = (const float*)d_in[10];

  char* ws = (char*)d_ws;
  size_t off = 0;
  auto alloc = [&](size_t bytes) -> void* {
    void* p = ws + off;
    off += (bytes + 255) & ~(size_t)255;
    return p;
  };
  bf16* xb  = (bf16*)alloc((size_t)ROWS_ * DIM_ * 2);
  bf16* wqb = (bf16*)alloc((size_t)(H_*QKD_) * DIM_ * 2);
  bf16* wab = (bf16*)alloc((size_t)(KVR_+ROPE_) * DIM_ * 2);
  bf16* wbb = (bf16*)alloc((size_t)(H_*(NOPE_+VDIM_)) * KVR_ * 2);
  bf16* wob = (bf16*)alloc((size_t)DIM_ * (H_*VDIM_) * 2);
  bf16* qb  = (bf16*)alloc((size_t)ROWS_ * (H_*QKD_) * 2);
  bf16* kva = (bf16*)alloc((size_t)ROWS_ * (KVR_+ROPE_) * 2);
  bf16* kvn = (bf16*)alloc((size_t)ROWS_ * KVR_ * 2);
  bf16* kb  = (bf16*)alloc((size_t)ROWS_ * (H_*QKD_) * 2);
  bf16* vb  = (bf16*)alloc((size_t)ROWS_ * (H_*VDIM_) * 2);
  bf16* ab  = (bf16*)alloc((size_t)ROWS_ * (H_*VDIM_) * 2);

  dim3 blk(256);
  auto cvt = [&](const float* src, bf16* dst, size_t n) {
    cvt_f32_bf16<<<dim3((unsigned)((n + 255) / 256)), blk, 0, stream>>>(src, dst, (int)n);
  };
  cvt(x,       xb,  (size_t)ROWS_ * DIM_);
  cvt(wq_w,    wqb, (size_t)(H_*QKD_) * DIM_);
  cvt(wkv_a_w, wab, (size_t)(KVR_+ROPE_) * DIM_);
  cvt(wkv_b_w, wbb, (size_t)(H_*(NOPE_+VDIM_)) * KVR_);
  cvt(wo_w,    wob, (size_t)DIM_ * (H_*VDIM_));

  // q = x @ wq^T + b      (M=8192, N=3072, K=1024) -> bf16
  gemm_wmma128<<<dim3((H_*QKD_)/128, ROWS_/128), blk, 0, stream>>>(
      xb, wqb, wq_b, qb, nullptr, ROWS_, H_*QKD_, DIM_, 0);
  // kv_a = x @ wkv_a^T + b (M=8192, N=576, K=1024) -> bf16 (N not /128 -> 64-tile kernel)
  gemm_wmma<<<dim3((KVR_+ROPE_)/64, ROWS_/64), blk, 0, stream>>>(
      xb, wab, wkv_a_b, kva, nullptr, ROWS_, KVR_+ROPE_, DIM_, 0);

  rope_q<<<dim3((ROWS_*H_*(ROPE_/2) + 255)/256), blk, 0, stream>>>(qb, fcos, fsin);
  kv_post<<<dim3(ROWS_), blk, 0, stream>>>(kva, kvn, kb, fcos, fsin);

  // kv = rms(kv_lat) @ wkv_b^T + b, scattered into K(nope)/V (M=8192, N=4096, K=512)
  gemm_wmma128<<<dim3((H_*(NOPE_+VDIM_))/128, ROWS_/128), blk, 0, stream>>>(
      kvn, wbb, wkv_b_b, kb, vb, ROWS_, H_*(NOPE_+VDIM_), KVR_, 2);

  // flash attention
  attn_wmma<<<dim3(T_/128, H_, B_), blk, 0, stream>>>(qb, kb, vb, ab);

  // out = attn @ wo^T + b  (M=8192, N=1024, K=2048) -> f32 d_out
  gemm_wmma128<<<dim3(DIM_/128, ROWS_/128), blk, 0, stream>>>(
      ab, wob, wo_b, d_out, nullptr, ROWS_, DIM_, H_*VDIM_, 1);
}
